// GraphAttentionLayer_58780922413922
// MI455X (gfx1250) — matrix operand order repaired and checked by reading, not compiled, not run on hardware
//
#include <hip/hip_runtime.h>
#include <hip/hip_bf16.h>

typedef __attribute__((ext_vector_type(16))) _Float16 v16h;
typedef __attribute__((ext_vector_type(8)))  _Float16 v8h;
typedef __attribute__((ext_vector_type(8)))  float    v8f;
typedef __attribute__((ext_vector_type(4)))  int      v4i;

#define N_ROWS 8192
#define IN_F   128
#define OUT_F  64
#define ALPHA  0.2f

// ---------------------------------------------------------------------------
// Kernel 1: Wh = h @ W  (f32), also emit WhT (f16, transposed [64][8192])
// ---------------------------------------------------------------------------
__global__ __launch_bounds__(256) void wh_kernel(
    const float* __restrict__ h, const float* __restrict__ W,
    float* __restrict__ Wh, _Float16* __restrict__ WhT)
{
    int idx = blockIdx.x * 256 + threadIdx.x;        // 8192*64 threads
    int row = idx >> 6;
    int col = idx & 63;
    const float* hr = h + (size_t)row * IN_F;
    float acc = 0.f;
#pragma unroll 8
    for (int k = 0; k < IN_F; ++k)
        acc += hr[k] * W[k * OUT_F + col];
    Wh[idx] = acc;
    WhT[(size_t)col * N_ROWS + row] = (_Float16)acc;
}

// ---------------------------------------------------------------------------
// Kernel 2: Wh1[i] = Wh[i,:]·a[0:64], Wh2[i] = Wh[i,:]·a[64:128]
// ---------------------------------------------------------------------------
__global__ __launch_bounds__(256) void wh12_kernel(
    const float* __restrict__ Wh, const float* __restrict__ a,
    float* __restrict__ Wh1, float* __restrict__ Wh2)
{
    int row = blockIdx.x * 256 + threadIdx.x;        // 8192 threads
    const float* r = Wh + (size_t)row * OUT_F;
    float s1 = 0.f, s2 = 0.f;
#pragma unroll 8
    for (int k = 0; k < OUT_F; ++k) {
        float v = r[k];
        s1 += v * a[k];
        s2 += v * a[OUT_F + k];
    }
    Wh1[row] = s1;
    Wh2[row] = s2;
}

// ---------------------------------------------------------------------------
// Kernel 3: fused masked-attention + softmax + (P @ Wh) / rowsum(P)
// Block = 256 threads (8 waves); block owns a 16-row output tile.
// adj is a 256MB read-once stream -> non-temporal loads so the 192MB L2
// stays dedicated to the heavily re-read WhT / Wh2 working set.
// ---------------------------------------------------------------------------
__global__ __launch_bounds__(256) void gat_kernel(
    const int*      __restrict__ adj,
    const float*    __restrict__ Wh1,
    const float*    __restrict__ Wh2,
    const _Float16* __restrict__ WhT,
    float*          __restrict__ out)
{
    __shared__ float accs[8][4][256];   // [wave][tile][M*16+n]  32 KB
    __shared__ float dls[8][32];        // per-lane denominator partials 1 KB

    const int lane = threadIdx.x & 31;
    const int wave = threadIdx.x >> 5;
    const int grp  = lane >> 4;          // 0: lanes 0-15, 1: lanes 16-31
    const int m    = lane & 15;          // row within A tile, col within B/C tile
    const int i0   = blockIdx.x * 16;
    const int row  = i0 + m;

    const float w1     = Wh1[row];
    const float NEGINF = -__builtin_inff();
    const int kb0 = grp * 8;             // A-operand K runs per ISA layout
    const int kb1 = 16 + grp * 8;

    const int* adjrow = adj + (size_t)row * N_ROWS;

    v8f c0 = {}, c1 = {}, c2 = {}, c3 = {};
    float dacc[4] = {0.f, 0.f, 0.f, 0.f};

    // 8192 cols / 32 per chunk / 8 waves = 32 iterations per wave
    for (int it = 0; it < 32; ++it) {
        const int j0 = (it * 8 + wave) * 32;

        // --- adjacency: read-once stream, non-temporal (4x b128 NT) ---
        v4i a0 = __builtin_nontemporal_load((const v4i*)(adjrow + j0 + kb0));
        v4i a1 = __builtin_nontemporal_load((const v4i*)(adjrow + j0 + kb0 + 4));
        v4i a2 = __builtin_nontemporal_load((const v4i*)(adjrow + j0 + kb1));
        v4i a3 = __builtin_nontemporal_load((const v4i*)(adjrow + j0 + kb1 + 4));

        // --- Wh2 values (hot, cacheable) ---
        float4 f0 = *(const float4*)(Wh2 + j0 + kb0);
        float4 f1 = *(const float4*)(Wh2 + j0 + kb0 + 4);
        float4 f2 = *(const float4*)(Wh2 + j0 + kb1);
        float4 f3 = *(const float4*)(Wh2 + j0 + kb1 + 4);

        int   av[16] = { a0[0],a0[1],a0[2],a0[3], a1[0],a1[1],a1[2],a1[3],
                         a2[0],a2[1],a2[2],a2[3], a3[0],a3[1],a3[2],a3[3] };
        float fv[16] = { f0.x,f0.y,f0.z,f0.w, f1.x,f1.y,f1.z,f1.w,
                         f2.x,f2.y,f2.z,f2.w, f3.x,f3.y,f3.z,f3.w };

        union { v16h v; _Float16 e[16]; } A;
#pragma unroll
        for (int q = 0; q < 16; ++q) {
            float e = w1 + fv[q];
            e = (e > 0.f) ? e : ALPHA * e;       // leaky relu (v_cndmask)
            e = (av[q] > 0) ? e : NEGINF;        // mask folded into exp argument
            float p = __expf(e);                 // exp(-inf) == 0, branchless
            dacc[q & 3] += p;                    // 4 independent chains
            A.e[q] = (_Float16)p;
        }

        // --- B tiles: WhT[col][j0 + grp*8 .. +8) and [j0 + 16 + grp*8 .. +8) (hot) ---
        const size_t boff = (size_t)j0 + grp * 8;
        union { v16h v; v8h h[2]; } B0, B1, B2, B3;
        {
            const _Float16* p0 = WhT + (size_t)(0 * 16 + m) * N_ROWS + boff;
            const _Float16* p1 = WhT + (size_t)(1 * 16 + m) * N_ROWS + boff;
            const _Float16* p2 = WhT + (size_t)(2 * 16 + m) * N_ROWS + boff;
            const _Float16* p3 = WhT + (size_t)(3 * 16 + m) * N_ROWS + boff;
            B0.h[0] = *(const v8h*)(p0);  B0.h[1] = *(const v8h*)(p0 + 16);
            B1.h[0] = *(const v8h*)(p1);  B1.h[1] = *(const v8h*)(p1 + 16);
            B2.h[0] = *(const v8h*)(p2);  B2.h[1] = *(const v8h*)(p2 + 16);
            B3.h[0] = *(const v8h*)(p3);  B3.h[1] = *(const v8h*)(p3 + 16);
        }

        c0 = __builtin_amdgcn_wmma_f32_16x16x32_f16(false, A.v, false, B0.v, (short)0, c0, false, false);
        c1 = __builtin_amdgcn_wmma_f32_16x16x32_f16(false, A.v, false, B1.v, (short)0, c1, false, false);
        c2 = __builtin_amdgcn_wmma_f32_16x16x32_f16(false, A.v, false, B2.v, (short)0, c2, false, false);
        c3 = __builtin_amdgcn_wmma_f32_16x16x32_f16(false, A.v, false, B3.v, (short)0, c3, false, false);
    }

    const float dsum = (dacc[0] + dacc[1]) + (dacc[2] + dacc[3]);

    // --- spill per-wave partial tiles to LDS (C layout: M = r + 8*grp) ---
#pragma unroll
    for (int r = 0; r < 8; ++r) {
        int M = r + 8 * grp;
        accs[wave][0][M * 16 + m] = c0[r];
        accs[wave][1][M * 16 + m] = c1[r];
        accs[wave][2][M * 16 + m] = c2[r];
        accs[wave][3][M * 16 + m] = c3[r];
    }
    dls[wave][lane] = dsum;
    __syncthreads();

    // --- cross-wave reduce, normalize, write 16x64 tile (write-once, NT) ---
    for (int e = threadIdx.x; e < 16 * OUT_F; e += 256) {
        int M = e >> 6, col = e & 63;
        int t = col >> 4, n = col & 15;
        float den = 0.f, s = 0.f;
#pragma unroll
        for (int w = 0; w < 8; ++w) {
            den += dls[w][M] + dls[w][16 + M];
            s   += accs[w][t][M * 16 + n];
        }
        if (den == 0.f) den = 1.f;
        __builtin_nontemporal_store(s / den, out + (size_t)(i0 + M) * OUT_F + col);
    }
}

// ---------------------------------------------------------------------------
extern "C" void kernel_launch(void* const* d_in, const int* in_sizes, int n_in,
                              void* d_out, int out_size, void* d_ws, size_t ws_size,
                              hipStream_t stream)
{
    const float* h   = (const float*)d_in[0];   // [8192,128]
    const float* W   = (const float*)d_in[1];   // [128,64]
    const float* a   = (const float*)d_in[2];   // [128,1]
    const int*   adj = (const int*)d_in[3];     // [8192,8192]
    float* out = (float*)d_out;                 // [8192,64]

    char* ws = (char*)d_ws;
    float*    Wh   = (float*)(ws);                          // 2 MB
    _Float16* WhT  = (_Float16*)(ws + 2097152);             // 1 MB
    float*    Wh1  = (float*)(ws + 2097152 + 1048576);      // 32 KB
    float*    Wh2  = (float*)(ws + 2097152 + 1048576 + 32768);

    wh_kernel<<<(N_ROWS * OUT_F) / 256, 256, 0, stream>>>(h, W, Wh, WhT);
    wh12_kernel<<<N_ROWS / 256, 256, 0, stream>>>(Wh, a, Wh1, Wh2);
    gat_kernel<<<N_ROWS / 16, 256, 0, stream>>>(adj, Wh1, Wh2, WhT, out);
}
